// SimpleALIFRNN_30571577213348
// MI455X (gfx1250) — matrix operand order repaired and checked by reading, not compiled, not run on hardware
//
#include <hip/hip_runtime.h>
#include <cstdint>
#include <cstddef>

#define T_STEPS 250
#define BATCH   256
#define IN_DIM  700
#define XPAD    704          // x part padded to multiple of 32
#define HID     1024
#define ACOLS   (XPAD + HID) // 1728: concatenated [x | z] row length (bf16)
#define OUT_DIM 20
#define KCX     (XPAD / 32)  // 22
#define KCZ     (HID / 32)   // 32
#define KCT     (KCX + KCZ)  // 54
#define NTILES  (HID / 16)   // 64
#define MTILES  (BATCH / 16) // 16

typedef __attribute__((ext_vector_type(16))) __bf16         v16bf;
typedef __attribute__((ext_vector_type(8)))  float          v8f;
typedef __attribute__((ext_vector_type(16))) unsigned short v16u;

__device__ __forceinline__ unsigned short f2bf(float f) {
    unsigned int u = __builtin_bit_cast(unsigned int, f);
    unsigned int r = u + 0x7FFFu + ((u >> 16) & 1u);   // round-to-nearest-even
    return (unsigned short)(r >> 16);
}
__device__ __forceinline__ float bf2f(unsigned short h) {
    unsigned int u = ((unsigned int)h) << 16;
    return __builtin_bit_cast(float, u);
}
__device__ __forceinline__ v8f wmma_bf16(v16u a, v16u b, v8f c) {
    return __builtin_amdgcn_wmma_f32_16x16x32_bf16(
        false, __builtin_bit_cast(v16bf, a), false, __builtin_bit_cast(v16bf, b),
        (short)0, c, false, false);
}

// ---------------------------------------------------------------------------
// Init: zero states (uf/af/uof/n_spk live in d_out), precompute decay tables,
// and fill activation buffer 0 with [bf16(x_0) | 0-pad | z=0].
// ---------------------------------------------------------------------------
__global__ void alif_init(const float* __restrict__ x0,
                          const float* __restrict__ tau_mem,
                          const float* __restrict__ tau_adp,
                          const float* __restrict__ tau_out,
                          float* __restrict__ alpha, float* __restrict__ rho,
                          float* __restrict__ aout,
                          float* __restrict__ uf, float* __restrict__ af,
                          float* __restrict__ uof, float* __restrict__ nspk,
                          unsigned short* __restrict__ ab0) {
    int i = blockIdx.x * blockDim.x + threadIdx.x;
    if (i < BATCH * HID) { uf[i] = 0.f; af[i] = 0.f; }
    if (i < BATCH * ACOLS) {
        int b = i / ACOLS, c = i - b * ACOLS;
        float v = (c < IN_DIM) ? x0[(size_t)b * IN_DIM + c] : 0.f;  // pad + z=0
        ab0[i] = f2bf(v);
    }
    if (i < BATCH * OUT_DIM) uof[i] = 0.f;
    if (i < HID) {
        alpha[i] = expf(-1.0f / tau_mem[i]);
        rho[i]   = expf(-1.0f / tau_adp[i]);
    }
    if (i < OUT_DIM) aout[i] = expf(-1.0f / tau_out[i]);
    if (i == 0) *nspk = 0.f;
}

// ---------------------------------------------------------------------------
// Pre-pack w_hidden [1724,1024] f32 -> bf16 in WMMA B-operand layout, with a
// contiguous K order (k = 16*(lane>>4) + e within each 32-chunk), matching
// the A operand where each lane reads 16 *contiguous* bf16.
// pw[((ntile*KCT + kc)*512) + lane*16 + e]:
//   n  = ntile*16 + (lane&15)
//   lk = ((lane>>4)<<4) | e                        (contiguous K within chunk)
//   kc <  KCX: k = kc*32 + lk            (x rows, zero-pad k>=700)
//   kc >= KCX: k = 700 + (kc-KCX)*32 + lk     (z rows)
// ---------------------------------------------------------------------------
__global__ void alif_pack_w(const float* __restrict__ w,
                            unsigned short* __restrict__ pw) {
    int idx = blockIdx.x * blockDim.x + threadIdx.x;
    if (idx >= NTILES * KCT * 512) return;
    int e     = idx & 15;
    int lane  = (idx >> 4) & 31;
    int blk   = idx >> 9;            // ntile*KCT + kc
    int kc    = blk % KCT;
    int ntile = blk / KCT;
    int n     = ntile * 16 + (lane & 15);
    int lk    = ((lane >> 4) << 4) | e;
    float v = 0.f;
    if (kc < KCX) {
        int k = kc * 32 + lk;
        if (k < IN_DIM) v = w[(size_t)k * HID + n];
    } else {
        int k = IN_DIM + (kc - KCX) * 32 + lk;   // <= 1723
        v = w[(size_t)k * HID + n];
    }
    pw[idx] = f2bf(v);
}

// ---------------------------------------------------------------------------
// Fused per-timestep kernel: cur = [x_t, z_{t-1}] @ W via bf16 WMMA
// (f32 accum), software-pipelined (depth-2 prefetch of A and B into distinct
// registers so loads overlap WMMAs), then ALIF update in-register.
// 512 waves: wave = mtile*32 + ngroup; each wave: one 16-row M tile,
// two 16-col N tiles (A operand shared across 2 WMMAs per K chunk).
// ---------------------------------------------------------------------------
__global__ __launch_bounds__(128) void alif_step(
        const unsigned short* __restrict__ abuf,  // [B][ACOLS] bf16 [x_t | z]
        unsigned short*       __restrict__ bnext, // next buf (z section written)
        const unsigned short* __restrict__ pw,    // packed W (bf16)
        const float* __restrict__ alpha, const float* __restrict__ rho,
        float* __restrict__ uf, float* __restrict__ af,
        float* __restrict__ zf) {
    const int lane  = threadIdx.x & 31;
    const int wave  = blockIdx.x * (blockDim.x >> 5) + (threadIdx.x >> 5);
    const int mtile = wave >> 5;           // 0..15
    const int nt0   = (wave & 31) * 2;     // 0..62
    const int khalf = lane >> 4;           // 0|1
    const int mrow  = lane & 15;
    const int arow  = mtile * 16 + mrow;   // A-matrix row this lane supplies

    // thanks to the K-permutation, each lane's A chunk is one aligned 32B load
    const unsigned short* ar  = abuf + (size_t)arow * ACOLS + khalf * 16;
    const unsigned short* pw0 = pw + (size_t)(nt0    ) * KCT * 512 + lane * 16;
    const unsigned short* pw1 = pw + (size_t)(nt0 + 1) * KCT * 512 + lane * 16;

    v8f c0 = {}; v8f c1 = {};

    // software pipeline, depth 2 (prefetch kc+2/kc+3 while computing kc/kc+1)
    v16u a0  = *(const v16u*)(ar + 0 * 32);
    v16u a1  = *(const v16u*)(ar + 1 * 32);
    v16u p00 = *(const v16u*)(pw0 + (size_t)0 * 512);
    v16u p10 = *(const v16u*)(pw1 + (size_t)0 * 512);
    v16u p01 = *(const v16u*)(pw0 + (size_t)1 * 512);
    v16u p11 = *(const v16u*)(pw1 + (size_t)1 * 512);

    for (int kc = 0; kc < KCT; kc += 2) {
        int n0 = (kc + 2 < KCT) ? kc + 2 : 0;   // tail: harmless re-read of 0
        int n1 = (kc + 3 < KCT) ? kc + 3 : 0;
        v16u na0 = *(const v16u*)(ar + n0 * 32);
        v16u na1 = *(const v16u*)(ar + n1 * 32);
        v16u q00 = *(const v16u*)(pw0 + (size_t)n0 * 512);
        v16u q10 = *(const v16u*)(pw1 + (size_t)n0 * 512);
        v16u q01 = *(const v16u*)(pw0 + (size_t)n1 * 512);
        v16u q11 = *(const v16u*)(pw1 + (size_t)n1 * 512);

        c0 = wmma_bf16(a0, p00, c0);
        c1 = wmma_bf16(a0, p10, c1);
        c0 = wmma_bf16(a1, p01, c0);
        c1 = wmma_bf16(a1, p11, c1);

        a0 = na0; a1 = na1;
        p00 = q00; p10 = q10; p01 = q01; p11 = q11;
    }

    // ---- fused ALIF epilogue on the two C tiles ----
    // C layout: element i of v8f -> row m = khalf*8 + i, col n = lane&15
    #pragma unroll
    for (int tile = 0; tile < 2; ++tile) {
        v8f cc = tile ? c1 : c0;
        int hcol = (nt0 + tile) * 16 + mrow;
        float al = alpha[hcol], rh = rho[hcol];
        #pragma unroll
        for (int i = 0; i < 8; ++i) {
            int brow = mtile * 16 + khalf * 8 + i;
            int idx  = brow * HID + hcol;
            float cur = cc[i];
            float aa  = af[idx];
            float uu  = uf[idx];
            float zp  = bf2f(abuf[(size_t)brow * ACOLS + XPAD + hcol]);
            aa = rh * aa + (1.f - rh) * zp;               // threshold adaptation
            float theta = 0.01f + 1.8f * aa;              // B0 + BETA*a
            uu = al * uu + (1.f - al) * cur - zp * theta; // membrane, soft reset
            float zn = (uu - theta) > 0.f ? 1.f : 0.f;    // spike
            af[idx] = aa;
            uf[idx] = uu;
            zf[idx] = zn;
            bnext[(size_t)brow * ACOLS + XPAD + hcol] = f2bf(zn); // exact in bf16
        }
    }
}

// ---------------------------------------------------------------------------
// Readout: u_out = aout*u_out + (1-aout)*(z_t @ w_out); outputs[t]=u_out;
// n_spk += sum(z_t). Also converts x_{t+1} f32->bf16 into the next
// activation buffer (off the WMMA kernel's critical loop).
// One block per batch row, 128 threads.
// ---------------------------------------------------------------------------
__global__ __launch_bounds__(128) void alif_readout(
        unsigned short* __restrict__ bnext,   // [B][ACOLS]: z_t at +XPAD; x dest
        const float* __restrict__ x_next,     // [B, IN_DIM] f32 (x_{t+1})
        const float* __restrict__ w_out,      // [HID, OUT_DIM]
        const float* __restrict__ aout,       // [OUT_DIM]
        float* __restrict__ uof,              // [BATCH, OUT_DIM] state
        float* __restrict__ out_t,            // outputs + t*B*O
        float* __restrict__ n_spk) {
    __shared__ float red[128 * 21];
    const int b = blockIdx.x, tid = threadIdx.x;

    // convert next step's x row into the activation buffer (pad 700..703 = 0)
    unsigned short* xrow = bnext + (size_t)b * ACOLS;
    const float*    xsrc = x_next + (size_t)b * IN_DIM;
    for (int c = tid; c < XPAD; c += 128) {
        float v = (c < IN_DIM) ? xsrc[c] : 0.f;
        xrow[c] = f2bf(v);
    }

    float acc[OUT_DIM];
    #pragma unroll
    for (int o = 0; o < OUT_DIM; ++o) acc[o] = 0.f;
    float zs = 0.f;
    const unsigned short* zrow = bnext + (size_t)b * ACOLS + XPAD;
    #pragma unroll
    for (int j = 0; j < 8; ++j) {
        int h = tid * 8 + j;
        float zv = bf2f(zrow[h]);
        zs += zv;
        if (zv != 0.f) {                       // z in {0,1}
            const float* wr = w_out + (size_t)h * OUT_DIM;
            #pragma unroll
            for (int o = 0; o < OUT_DIM; ++o) acc[o] += wr[o];
        }
    }
    #pragma unroll
    for (int o = 0; o < OUT_DIM; ++o) red[tid * 21 + o] = acc[o];
    red[tid * 21 + 20] = zs;
    __syncthreads();
    for (int s = 64; s > 0; s >>= 1) {
        if (tid < s) {
            #pragma unroll
            for (int k = 0; k < 21; ++k)
                red[tid * 21 + k] += red[(tid + s) * 21 + k];
        }
        __syncthreads();
    }
    if (tid < OUT_DIM) {
        float dot = red[tid];
        float ao  = aout[tid];
        float u   = uof[b * OUT_DIM + tid];
        u = ao * u + (1.f - ao) * dot;
        uof[b * OUT_DIM + tid] = u;
        out_t[b * OUT_DIM + tid] = u;
    }
    if (tid == 0) atomicAdd(n_spk, red[20]);
}

// ---------------------------------------------------------------------------
extern "C" void kernel_launch(void* const* d_in, const int* in_sizes, int n_in,
                              void* d_out, int out_size, void* d_ws, size_t ws_size,
                              hipStream_t stream) {
    const float* x        = (const float*)d_in[0];
    const float* w_hidden = (const float*)d_in[1];
    const float* tau_mem  = (const float*)d_in[2];
    const float* tau_adp  = (const float*)d_in[3];
    const float* w_out    = (const float*)d_in[4];
    const float* tau_out  = (const float*)d_in[5];

    // d_out flat layout: outputs[T,B,O], zf[B,H], uf[B,H], af[B,H], uof[B,O], n_spk
    float* out     = (float*)d_out;
    float* outputs = out;
    float* zf      = outputs + (size_t)T_STEPS * BATCH * OUT_DIM;
    float* uf      = zf  + (size_t)BATCH * HID;
    float* af      = uf  + (size_t)BATCH * HID;
    float* uof     = af  + (size_t)BATCH * HID;
    float* nspk    = uof + (size_t)BATCH * OUT_DIM;

    // ws layout (~5.4 MB): packed W bf16, double-buffered [x|z] bf16, tables
    const size_t PWE = (size_t)NTILES * KCT * 512;       // 1,769,472 bf16
    const size_t ABE = (size_t)BATCH * ACOLS;            //   442,368 bf16 each
    unsigned short* pw  = (unsigned short*)d_ws;
    unsigned short* ab0 = pw + PWE;
    unsigned short* ab1 = ab0 + ABE;
    float* alpha = (float*)(ab1 + ABE);
    float* rho   = alpha + HID;
    float* aout  = rho + HID;

    alif_init<<<(int)((BATCH * ACOLS + 255) / 256), 256, 0, stream>>>(
        x, tau_mem, tau_adp, tau_out, alpha, rho, aout, uf, af, uof, nspk, ab0);
    alif_pack_w<<<(int)((PWE + 255) / 256), 256, 0, stream>>>(w_hidden, pw);

    unsigned short* ab[2] = { ab0, ab1 };
    for (int t = 0; t < T_STEPS; ++t) {
        unsigned short* cur = ab[t & 1];
        unsigned short* nxt = ab[(t + 1) & 1];
        alif_step<<<(MTILES * 32) / 4, 128, 0, stream>>>(
            cur, nxt, pw, alpha, rho, uf, af, zf);
        int tn = (t + 1 < T_STEPS) ? (t + 1) : (T_STEPS - 1); // harmless clamp
        alif_readout<<<BATCH, 128, 0, stream>>>(
            nxt, x + (size_t)tn * BATCH * IN_DIM, w_out, aout, uof,
            outputs + (size_t)t * BATCH * OUT_DIM, nspk);
        (void)in_sizes; (void)n_in; (void)out_size; (void)ws_size;
    }
}
